// LiquidCfCRouter_51531017617707
// MI455X (gfx1250) — matrix operand order repaired and checked by reading, not compiled, not run on hardware
//
#include <hip/hip_runtime.h>

#define HID    512
#define NEXP   128
#define BATCH  4
#define SEQ    2048
#define NWG    16
#define WSL    32      // cols of each 512-wide weight owned per WG
#define GSL    64      // cols of w_g1 (1024 wide) owned per WG
#define SMOOTH_F 0.1f

typedef _Float16 v16h __attribute__((ext_vector_type(16)));
typedef float    v8f  __attribute__((ext_vector_type(8)));

union Frag { v16h v; uint4 q[2]; };

__device__ __forceinline__ v8f wmma32(const Frag& a, const Frag& b, v8f c) {
  return __builtin_amdgcn_wmma_f32_16x16x32_f16(false, a.v, false, b.v, (short)0, c, false, false);
}

// A fragment: 16x32 f16, row-major, base already offset to k0. ISA 7.12.2 layout:
// lane L: m=L%15.., halves 0..7 -> K = (L>=16?8:0)+0..7 ; halves 8..15 -> +16.
__device__ __forceinline__ Frag frag_a(const _Float16* base, int ld, int lane) {
  Frag f; int m = lane & 15, h8 = (lane >> 4) * 8;
  const _Float16* p = base + m * ld + h8;
  f.q[0] = *(const uint4*)p; f.q[1] = *(const uint4*)(p + 16);
  return f;
}
// B fragment: 32x16 f16 stored TRANSPOSED (N-major rows, stride ld):
// lane L: n=L%16, halves 0..7 -> K = (L>=16?8:0)+0..7 ; halves 8..15 -> +16.
__device__ __forceinline__ Frag frag_b(const _Float16* base, int ld, int lane) {
  Frag f; int n = lane & 15, hk = (lane >> 4) * 8;
  const _Float16* p = base + n * ld + hk;
  f.q[0] = *(const uint4*)p; f.q[1] = *(const uint4*)(p + 16);
  return f;
}

// one 16x16 output tile, A(16xK) row-major stride lda, B^T rows at `B` stride ldb
__device__ __forceinline__ v8f gemm16(const _Float16* A, int lda,
                                      const _Float16* B, int ldb, int K, int lane) {
  v8f acc = {};
  for (int k0 = 0; k0 < K; k0 += 32) {
    Frag a = frag_a(A + k0, lda, lane);
    Frag b = frag_b(B + k0, ldb, lane);
    acc = wmma32(a, b, acc);
  }
  return acc;
}

__device__ __forceinline__ float gelu_f(float x) {
  return 0.5f * x * (1.0f + erff(x * 0.70710678118654752440f));
}

// ---------------- prep: f32 (K,N) -> f16 transposed (N,K) -------------------
__global__ __launch_bounds__(256) void kconv_t(const float* __restrict__ src,
                                               _Float16* __restrict__ dst,
                                               int K, int N, int rowOff) {
  int i = blockIdx.x * 256 + threadIdx.x;
  if (i < K * N) {
    int n = i / K, k = i - n * K;
    dst[i] = (_Float16)src[(size_t)(k + rowOff) * N + n];
  }
}

__global__ void kzero(unsigned* p, int n) {
  int i = blockIdx.x * 64 + threadIdx.x;
  if (i < n) p[i] = 0u;
}

// ---------------- phase 1a: XT = f16( x(f32) @ Win + b ) --------------------
__global__ __launch_bounds__(256) void gemm_xin(const float* __restrict__ X,
                                                const _Float16* __restrict__ WT,
                                                const float* __restrict__ bias,
                                                _Float16* __restrict__ OUT,
                                                int M, int N, int K) {
  __shared__ __align__(16) _Float16 As[64][40];
  __shared__ __align__(16) _Float16 Bs[64][40];
  int tid = threadIdx.x, lane = tid & 31, wave = tid >> 5;
  int m0 = blockIdx.x * 64, n0 = blockIdx.y * 64;
  int mi = wave >> 1, nset = wave & 1;
  v8f acc0 = {}, acc1 = {};
  for (int k0 = 0; k0 < K; k0 += 32) {
    #pragma unroll
    for (int j = 0; j < 2; ++j) {            // A: 64x32 f32 -> f16
      int idx = tid + 256 * j;
      int row = idx >> 3, c4 = (idx & 7) * 4;
      float4 f = *(const float4*)&X[(size_t)(m0 + row) * K + k0 + c4];
      As[row][c4] = (_Float16)f.x; As[row][c4 + 1] = (_Float16)f.y;
      As[row][c4 + 2] = (_Float16)f.z; As[row][c4 + 3] = (_Float16)f.w;
    }
    { int row = tid >> 2, c8 = (tid & 3) * 8;  // B^T: 64x32 halves
      *(uint4*)&Bs[row][c8] = *(const uint4*)&WT[(size_t)(n0 + row) * K + k0 + c8]; }
    __syncthreads();
    Frag a  = frag_a(&As[mi * 16][0], 40, lane);
    Frag b0 = frag_b(&Bs[nset * 32][0], 40, lane);
    Frag b1 = frag_b(&Bs[nset * 32 + 16][0], 40, lane);
    acc0 = wmma32(a, b0, acc0);
    acc1 = wmma32(a, b1, acc1);
    __syncthreads();
  }
  int n = lane & 15, hi = lane >> 4;
  #pragma unroll
  for (int r = 0; r < 8; ++r) {
    int m = m0 + mi * 16 + r + 8 * hi;
    int c0 = n0 + nset * 32 + n;
    OUT[(size_t)m * N + c0]      = (_Float16)(acc0[r] + bias[c0]);
    OUT[(size_t)m * N + c0 + 16] = (_Float16)(acc1[r] + bias[c0 + 16]);
  }
}

// ---------------- phase 1b: OUT(f32) = [gelu](XT @ WT + b1 [+ b2]) ----------
__global__ __launch_bounds__(256) void gemm_hid(const _Float16* __restrict__ A,
                                                const _Float16* __restrict__ WT,
                                                const float* __restrict__ b1,
                                                const float* __restrict__ b2,
                                                float* __restrict__ OUT,
                                                int M, int N, int K, int mode) {
  __shared__ __align__(16) _Float16 As[64][40];
  __shared__ __align__(16) _Float16 Bs[64][40];
  int tid = threadIdx.x, lane = tid & 31, wave = tid >> 5;
  int m0 = blockIdx.x * 64, n0 = blockIdx.y * 64;
  int mi = wave >> 1, nset = wave & 1;
  v8f acc0 = {}, acc1 = {};
  for (int k0 = 0; k0 < K; k0 += 32) {
    { int row = tid >> 2, c8 = (tid & 3) * 8;
      *(uint4*)&As[row][c8] = *(const uint4*)&A[(size_t)(m0 + row) * K + k0 + c8];
      *(uint4*)&Bs[row][c8] = *(const uint4*)&WT[(size_t)(n0 + row) * K + k0 + c8]; }
    __syncthreads();
    Frag a  = frag_a(&As[mi * 16][0], 40, lane);
    Frag b0 = frag_b(&Bs[nset * 32][0], 40, lane);
    Frag b1 = frag_b(&Bs[nset * 32 + 16][0], 40, lane);
    acc0 = wmma32(a, b0, acc0);
    acc1 = wmma32(a, b1, acc1);
    __syncthreads();
  }
  int n = lane & 15, hi = lane >> 4;
  #pragma unroll
  for (int r = 0; r < 8; ++r) {
    int m = m0 + mi * 16 + r + 8 * hi;
    #pragma unroll
    for (int s = 0; s < 2; ++s) {
      int c = n0 + nset * 32 + n + 16 * s;
      float v = (s ? acc1[r] : acc0[r]) + b1[c];
      if (mode) v = gelu_f(v + b2[c]);
      OUT[(size_t)m * N + c] = v;
    }
  }
}

// ---------------- inter-workgroup barrier -----------------------------------
// Cluster dispatch -> s_barrier_signal/wait -3 (one signal per WG, wave0).
// Plain dispatch (cluster_id==0) -> L2 atomic sense barrier.
__device__ __forceinline__ void grid_sync(unsigned* bar, unsigned* gen) {
  __syncthreads();
  if (__builtin_amdgcn_cluster_id_x() != 0) {
    if (threadIdx.x < 32) __builtin_amdgcn_s_cluster_barrier();
  } else if (threadIdx.x == 0) {
    __threadfence();
    unsigned g = ++(*gen);
    atomicAdd(bar, 1u);
    while (atomicAdd(bar, 0u) < g * (unsigned)NWG) __builtin_amdgcn_s_sleep(1);
    __threadfence();
  }
  __syncthreads();
}

__device__ __forceinline__ void copy_h8(_Float16* dst, const _Float16* src, int nh, int tid) {
  for (int i = tid * 8; i < nh; i += 256 * 8)
    *(uint4*)&dst[i] = *(const uint4*)&src[i];
}

// ---------------- phase 2: persistent 16-WG recurrent scan ------------------
__global__ __launch_bounds__(256) void liquid_scan(
    const float* __restrict__ preff1, const float* __restrict__ preiw,
    const _Float16* __restrict__ wT_ff1b, const _Float16* __restrict__ wT_ff2,
    const _Float16* __restrict__ wT_ta,   const _Float16* __restrict__ wT_tb,
    const _Float16* __restrict__ wT_rw,   const _Float16* __restrict__ wT_g1,
    const _Float16* __restrict__ wT_g2,
    const float* __restrict__ b_ff2, const float* __restrict__ b_ta,
    const float* __restrict__ b_tb,  const float* __restrict__ r_b,
    const float* __restrict__ b_g1,  const float* __restrict__ b_g2,
    _Float16* ff1g, _Float16* ff2g, _Float16* hg, _Float16* gg,
    float* logits, unsigned* bar, float* out) {
  extern __shared__ __align__(16) char smem[];
  _Float16* s_ff1b = (_Float16*)smem;            // 32x512
  _Float16* s_ff2  = s_ff1b + WSL * HID;
  _Float16* s_ta   = s_ff2  + WSL * HID;
  _Float16* s_tb   = s_ta   + WSL * HID;
  _Float16* s_rw   = s_tb   + WSL * HID;
  _Float16* s_g1   = s_rw   + WSL * HID;         // 64x512
  _Float16* hbuf   = s_g1   + GSL * HID;         // 16x512 (rows 0..3 live)
  _Float16* tbuf   = hbuf   + 16 * HID;          // 16x1024 (ff1/ff2/g staging)
  float* sbuf  = (float*)(tbuf + 16 * 2 * HID);  // 3 x 4 x 32
  float* lbuf  = sbuf + 384;                     // 4 x 128 logits
  float* prevp = lbuf + 512;                     // 4 x 128 prev probs
  float* red   = prevp + 512;                    // 128 reduce
  int*   ridx  = (int*)(red + 128);              // 128 argmax

  const int tid = threadIdx.x, lane = tid & 31, wave = tid >> 5;
  const int wg = blockIdx.x;
  unsigned mygen = 0;

  // pin this WG's weight slices in LDS (N-sliced, transposed/f16)
  copy_h8(s_ff1b, wT_ff1b + (size_t)wg * WSL * HID, WSL * HID, tid);
  copy_h8(s_ff2,  wT_ff2  + (size_t)wg * WSL * HID, WSL * HID, tid);
  copy_h8(s_ta,   wT_ta   + (size_t)wg * WSL * HID, WSL * HID, tid);
  copy_h8(s_tb,   wT_tb   + (size_t)wg * WSL * HID, WSL * HID, tid);
  copy_h8(s_rw,   wT_rw   + (size_t)wg * WSL * HID, WSL * HID, tid);
  copy_h8(s_g1,   wT_g1   + (size_t)wg * GSL * HID, GSL * HID, tid);
  for (int i = tid; i < 16 * HID; i += 256) hbuf[i] = (_Float16)0;      // h0 = 0 + M-pad
  for (int i = tid; i < 16 * 2 * HID; i += 256) tbuf[i] = (_Float16)0;  // M-pad rows
  for (int i = tid; i < 512; i += 256) prevp[i] = 0.f;
  __syncthreads();
  grid_sync(bar, &mygen);

  for (int t = 0; t < SEQ; ++t) {
    if (t + 1 < SEQ && tid < BATCH) {   // global_prefetch_b8 for next step's rows
      __builtin_prefetch(&preff1[((size_t)tid * SEQ + t + 1) * HID], 0, 1);
      __builtin_prefetch(&preiw [((size_t)tid * SEQ + t + 1) * HID], 0, 1);
    }
    // ---- A: ff1 = gelu(preff1 + h @ Wff1_bot) ----
    if (wave < 2) {
      v8f acc = gemm16(hbuf, HID, s_ff1b + wave * 16 * HID, HID, HID, lane);
      if (lane < 16) {
        #pragma unroll
        for (int r = 0; r < BATCH; ++r) {
          int col = wg * WSL + wave * 16 + lane;
          float v = acc[r] + preff1[((size_t)r * SEQ + t) * HID + col];
          ff1g[r * HID + col] = (_Float16)gelu_f(v);
        }
      }
    }
    grid_sync(bar, &mygen);
    { int r = tid >> 6, c = (tid & 63) * 8;    // gather full ff1
      *(uint4*)&tbuf[r * 2 * HID + c] = *(const uint4*)&ff1g[r * HID + c]; }
    __syncthreads();

    // ---- B: ff2 = gelu(ff1 @ Wff2 + b_ff2) ----
    if (wave < 2) {
      v8f acc = gemm16(tbuf, 2 * HID, s_ff2 + wave * 16 * HID, HID, HID, lane);
      if (lane < 16) {
        #pragma unroll
        for (int r = 0; r < BATCH; ++r) {
          int col = wg * WSL + wave * 16 + lane;
          ff2g[r * HID + col] = (_Float16)gelu_f(acc[r] + b_ff2[col]);
        }
      }
    }
    grid_sync(bar, &mygen);
    { int r = tid >> 6, c = (tid & 63) * 8;    // gather full ff2
      *(uint4*)&tbuf[r * 2 * HID + c] = *(const uint4*)&ff2g[r * HID + c]; }
    __syncthreads();

    // ---- C: ta/tb (from ff2), rec (from h) -> new_h slice ----
    if (wave < 6) {
      int which = wave >> 1, sub = wave & 1;
      const _Float16* W = (which == 0) ? s_ta : (which == 1) ? s_tb : s_rw;
      const _Float16* Ab = (which == 2) ? hbuf : tbuf;
      int lda = (which == 2) ? HID : 2 * HID;
      v8f acc = gemm16(Ab, lda, W + sub * 16 * HID, HID, HID, lane);
      if (lane < 16) {
        #pragma unroll
        for (int r = 0; r < BATCH; ++r) {
          int c = sub * 16 + lane, col = wg * WSL + c;
          float v = acc[r];
          if (which == 0)      v += b_ta[col];
          else if (which == 1) v += b_tb[col];
          else                 v = gelu_f(v + r_b[col]);
          sbuf[which * 128 + r * 32 + c] = v;
        }
      }
    }
    __syncthreads();
    if (tid < 128) {
      int r = tid >> 5, c = tid & 31, col = wg * WSL + c;
      float ti = 1.f / (1.f + __expf(-(sbuf[r * 32 + c] + sbuf[128 + r * 32 + c])));
      float inp = preiw[((size_t)r * SEQ + t) * HID + col];
      float hold = (float)hbuf[r * HID + col];
      float nh = hold * (1.f - ti) + ti * (inp + sbuf[256 + r * 32 + c]);
      hg[r * HID + col] = (_Float16)nh;
    }
    grid_sync(bar, &mygen);
    { int r = tid >> 6, c = (tid & 63) * 8;    // gather full new_h
      *(uint4*)&hbuf[r * HID + c] = *(const uint4*)&hg[r * HID + c]; }
    __syncthreads();

    // ---- D: g = gelu(new_h @ Wg1 + b_g1) ----
    if (wave < 4) {
      v8f acc = gemm16(hbuf, HID, s_g1 + wave * 16 * HID, HID, HID, lane);
      if (lane < 16) {
        #pragma unroll
        for (int r = 0; r < BATCH; ++r) {
          int col = wg * GSL + wave * 16 + lane;
          gg[r * 2 * HID + col] = (_Float16)gelu_f(acc[r] + b_g1[col]);
        }
      }
    }
    grid_sync(bar, &mygen);
    #pragma unroll
    for (int j = 0; j < 2; ++j) {              // gather full g (4x1024)
      int idx = tid + 256 * j, r = idx >> 7, c = (idx & 127) * 8;
      *(uint4*)&tbuf[r * 2 * HID + c] = *(const uint4*)&gg[r * 2 * HID + c];
    }
    __syncthreads();

    // ---- E: logits = g @ Wg2 + b_g2 (WGs 0..7, B-frags stream from L2) ----
    if (wg < 8 && wave == 0) {
      v8f acc = gemm16(tbuf, 2 * HID, wT_g2 + (size_t)(wg * 16) * 2 * HID,
                       2 * HID, 2 * HID, lane);
      if (lane < 16) {
        #pragma unroll
        for (int r = 0; r < BATCH; ++r) {
          int col = wg * 16 + lane;
          logits[r * NEXP + col] = acc[r] + b_g2[col];
        }
      }
    }
    grid_sync(bar, &mygen);

    // ---- F: softmax + smoothing + top-2 (redundant per WG; WG0 writes) ----
    for (int i = tid; i < BATCH * NEXP; i += 256) lbuf[i] = logits[i];
    __syncthreads();
    for (int r = 0; r < BATCH; ++r) {
      float p = 0.f;
      if (tid < NEXP) red[tid] = lbuf[r * NEXP + tid];
      __syncthreads();
      for (int s = 64; s > 0; s >>= 1) { if (tid < s) red[tid] = fmaxf(red[tid], red[tid + s]); __syncthreads(); }
      float mx = red[0]; __syncthreads();
      if (tid < NEXP) { p = __expf(lbuf[r * NEXP + tid] - mx); red[tid] = p; }
      __syncthreads();
      for (int s = 64; s > 0; s >>= 1) { if (tid < s) red[tid] += red[tid + s]; __syncthreads(); }
      float inv = 1.f / red[0]; __syncthreads();
      if (tid < NEXP) {
        p *= inv;
        float ps = (t == 0) ? p : (1.f - SMOOTH_F) * p + SMOOTH_F * prevp[r * NEXP + tid];
        prevp[r * NEXP + tid] = ps;
        red[tid] = ps; ridx[tid] = tid;
      }
      __syncthreads();
      for (int s = 64; s > 0; s >>= 1) {
        if (tid < s && red[tid + s] > red[tid]) { red[tid] = red[tid + s]; ridx[tid] = ridx[tid + s]; }
        __syncthreads();
      }
      float v1 = red[0]; int i1 = ridx[0]; __syncthreads();
      if (tid < NEXP) { red[tid] = (tid == i1) ? -1e30f : prevp[r * NEXP + tid]; ridx[tid] = tid; }
      __syncthreads();
      for (int s = 64; s > 0; s >>= 1) {
        if (tid < s && red[tid + s] > red[tid]) { red[tid] = red[tid + s]; ridx[tid] = ridx[tid + s]; }
        __syncthreads();
      }
      float v2 = red[0]; int i2 = ridx[0]; __syncthreads();
      if (wg == 0 && tid == 0) {
        size_t base = ((size_t)r * SEQ + t) * 2;
        const size_t OUTOFF = (size_t)BATCH * SEQ * 2;
        float sw = v1 + v2;
        out[base] = (float)i1; out[base + 1] = (float)i2;
        out[OUTOFF + base] = v1 / sw; out[OUTOFF + base + 1] = v2 / sw;
      }
      __syncthreads();
    }
  }
}

// ---------------------------------------------------------------------------
extern "C" void kernel_launch(void* const* d_in, const int* in_sizes, int n_in,
                              void* d_out, int out_size, void* d_ws, size_t ws_size,
                              hipStream_t stream) {
  (void)in_sizes; (void)n_in; (void)out_size; (void)ws_size;
  const float* x        = (const float*)d_in[0];
  const float* w_in     = (const float*)d_in[1];
  const float* b_in     = (const float*)d_in[2];
  const float* w_ff1    = (const float*)d_in[3];
  const float* b_ff1    = (const float*)d_in[4];
  const float* w_ff2    = (const float*)d_in[5];
  const float* b_ff2    = (const float*)d_in[6];
  const float* w_ta     = (const float*)d_in[7];
  const float* b_ta     = (const float*)d_in[8];
  const float* w_tb     = (const float*)d_in[9];
  const float* b_tb     = (const float*)d_in[10];
  const float* w_iw     = (const float*)d_in[11];
  const float* b_iw     = (const float*)d_in[12];
  const float* input_b  = (const float*)d_in[13];
  const float* w_rw     = (const float*)d_in[14];
  const float* r_b      = (const float*)d_in[15];
  const float* w_g1     = (const float*)d_in[16];
  const float* b_g1     = (const float*)d_in[17];
  const float* w_g2     = (const float*)d_in[18];
  const float* b_g2     = (const float*)d_in[19];

  char* ws = (char*)d_ws; size_t off = 0;
  auto alloc = [&](size_t bytes) -> char* {
    off = (off + 255) & ~(size_t)255; char* p = ws + off; off += bytes; return p;
  };
  const int M = BATCH * SEQ;  // 8192 tokens
  _Float16* wT_in   = (_Float16*)alloc((size_t)HID * 4096 * 2);
  _Float16* wT_ff1t = (_Float16*)alloc((size_t)HID * HID * 2);
  _Float16* wT_ff1b = (_Float16*)alloc((size_t)HID * HID * 2);
  _Float16* wT_ff2  = (_Float16*)alloc((size_t)HID * HID * 2);
  _Float16* wT_ta   = (_Float16*)alloc((size_t)HID * HID * 2);
  _Float16* wT_tb   = (_Float16*)alloc((size_t)HID * HID * 2);
  _Float16* wT_iw   = (_Float16*)alloc((size_t)HID * HID * 2);
  _Float16* wT_rw   = (_Float16*)alloc((size_t)HID * HID * 2);
  _Float16* wT_g1   = (_Float16*)alloc((size_t)2 * HID * HID * 2);
  _Float16* wT_g2   = (_Float16*)alloc((size_t)NEXP * 2 * HID * 2);
  _Float16* XT      = (_Float16*)alloc((size_t)M * HID * 2);
  float*    preff1  = (float*)alloc((size_t)M * HID * 4);
  float*    preiw   = (float*)alloc((size_t)M * HID * 4);
  _Float16* ff1g    = (_Float16*)alloc(BATCH * HID * 2);
  _Float16* ff2g    = (_Float16*)alloc(BATCH * HID * 2);
  _Float16* hg      = (_Float16*)alloc(BATCH * HID * 2);
  _Float16* gg      = (_Float16*)alloc(BATCH * 2 * HID * 2);
  float*    logits  = (float*)alloc(BATCH * NEXP * 4);
  unsigned* bar     = (unsigned*)alloc(256);

  auto T = [&](const float* s, _Float16* d, int K, int N, int rowOff) {
    int tot = K * N;
    kconv_t<<<(tot + 255) / 256, 256, 0, stream>>>(s, d, K, N, rowOff);
  };
  T(w_in,  wT_in,   4096, HID, 0);
  T(w_ff1, wT_ff1t, HID,  HID, 0);
  T(w_ff1, wT_ff1b, HID,  HID, HID);
  T(w_ff2, wT_ff2,  HID,  HID, 0);
  T(w_ta,  wT_ta,   HID,  HID, 0);
  T(w_tb,  wT_tb,   HID,  HID, 0);
  T(w_iw,  wT_iw,   HID,  HID, 0);
  T(w_rw,  wT_rw,   HID,  HID, 0);
  T(w_g1,  wT_g1,   HID,  2 * HID, 0);
  T(w_g2,  wT_g2,   2 * HID, NEXP, 0);

  dim3 g1g(M / 64, HID / 64);
  gemm_xin<<<g1g, 256, 0, stream>>>(x, wT_in, b_in, XT, M, HID, 4096);
  gemm_hid<<<g1g, 256, 0, stream>>>(XT, wT_ff1t, b_ff1, b_ff1, preff1, M, HID, HID, 0);
  gemm_hid<<<g1g, 256, 0, stream>>>(XT, wT_iw,   b_iw, input_b, preiw, M, HID, HID, 1);

  kzero<<<1, 64, 0, stream>>>(bar, 64);

  size_t smem2 = (size_t)(5 * WSL * HID + GSL * HID + 16 * HID + 16 * 2 * HID) * 2
               + (384 + 512 + 512 + 128) * 4 + 128 * 4;   // ~278.5 KB of 320 KB
  liquid_scan<<<NWG, 256, smem2, stream>>>(
      preff1, preiw, wT_ff1b, wT_ff2, wT_ta, wT_tb, wT_rw, wT_g1, wT_g2,
      b_ff2, b_ta, b_tb, r_b, b_g1, b_g2,
      ff1g, ff2g, hg, gg, logits, bar, (float*)d_out);
}
